// GraphModel_35244501631244
// MI455X (gfx1250) — compile-verified, matrix operand order checked by reading, not run on hardware
//
#include <hip/hip_runtime.h>

typedef __attribute__((ext_vector_type(16))) _Float16 v16h;
typedef __attribute__((ext_vector_type(8)))  _Float16 v8h;
typedef __attribute__((ext_vector_type(8)))  float    v8f;

constexpr int B = 8, T = 2048, D = 768;
constexpr int NT = T / 16;          // 128 row tiles
constexpr int MT = 8;               // d-tiles per wave
constexpr int DG = (D / 16) / MT;   // 6 d-groups (6 * 128 cols = 768)

// workspace layout (bytes)
constexpr size_t OFF_EMBT = 0;                                   // _Float16 [B][D][T]  (25.2 MB)
constexpr size_t OFF_M    = OFF_EMBT + (size_t)B * D * T * 2;    // _Float16 [B][T][T]  (67.1 MB)

// 1) embedding gather -> full f32 output (already correct for non-node rows)
__global__ void k_embed(const int* __restrict__ code, const float* __restrict__ W,
                        float* __restrict__ out) {
    int row = blockIdx.x;                       // b*T + t
    const float4* src = (const float4*)(W + (size_t)code[row] * D);
    float4* dst = (float4*)(out + (size_t)row * D);
    dst[threadIdx.x] = src[threadIdx.x];        // 192 threads * float4 = 768 floats
}

// 2) row counts + materialize row-normalized mask m as f16 (zeros for non-node rows)
__global__ void k_cnt_m(const int* __restrict__ pos, const unsigned char* __restrict__ attn,
                        _Float16* __restrict__ m) {
    int row = blockIdx.x;                       // b*T + n
    int b = row >> 11;                          // / T
    int t8 = threadIdx.x * 8;                   // 256 threads * 8 = T
    _Float16* mrow = m + (size_t)row * T;
    v8h o;
    if (pos[row] != 0) {                        // block-uniform: non-node row -> zero
        #pragma unroll
        for (int j = 0; j < 8; ++j) o[j] = (_Float16)0.0f;
        *(v8h*)(mrow + t8) = o;
        return;
    }
    const unsigned char* ar = attn + (size_t)row * T;
    const int* pr = pos + b * T;
    bool on[8];
    int cnt = 0;
    #pragma unroll
    for (int j = 0; j < 8; ++j) {
        on[j] = (ar[t8 + j] != 0) && (pr[t8 + j] >= 2);
        cnt += on[j] ? 1 : 0;
    }
    __shared__ int red[256];
    red[threadIdx.x] = cnt;
    __syncthreads();
    for (int s = 128; s > 0; s >>= 1) {
        if ((int)threadIdx.x < s) red[threadIdx.x] += red[threadIdx.x + s];
        __syncthreads();
    }
    _Float16 ih = (_Float16)(1.0f / ((float)red[0] + 1e-10f));
    #pragma unroll
    for (int j = 0; j < 8; ++j) o[j] = on[j] ? ih : (_Float16)0.0f;
    *(v8h*)(mrow + t8) = o;
}

// 3) LDS-tiled transpose + f32->f16: embT[b][d][t]
__global__ void k_transpose(const float* __restrict__ out, _Float16* __restrict__ embT) {
    __shared__ float tile[32][33];
    int b = blockIdx.z, t0 = blockIdx.y * 32, d0 = blockIdx.x * 32;
    int lx = threadIdx.x & 31, ly = threadIdx.x >> 5;   // 32 x 8
    #pragma unroll
    for (int i = 0; i < 32; i += 8)
        tile[ly + i][lx] = out[((size_t)b * T + t0 + ly + i) * D + d0 + lx];
    __syncthreads();
    #pragma unroll
    for (int i = 0; i < 32; i += 8)
        embT[((size_t)b * D + d0 + ly + i) * T + t0 + lx] = (_Float16)tile[lx][ly + i];
}

// 4) WMMA einsum + blend: one wave = 16 node-rows x 128 cols, pure load+wmma inner loop
__global__ void k_gemm(const int* __restrict__ pos, const _Float16* __restrict__ m,
                       const _Float16* __restrict__ embT, float* __restrict__ out) {
    int wave = threadIdx.x >> 5;
    int lane = threadIdx.x & 31;
    int tile = blockIdx.x * (blockDim.x >> 5) + wave;
    int dG = tile % DG;
    int nT = (tile / DG) % NT;
    int b  = tile / (DG * NT);
    int nBase = nT * 16;
    int dBase = dG * (16 * MT);

    // wave-uniform early exit: no node rows in this 16-row tile
    bool any = false;
    #pragma unroll
    for (int r = 0; r < 16; ++r) any |= (pos[b * T + nBase + r] == 0);
    if (!any) return;

    int rl = lane & 15;
    int kb = (lane < 16) ? 0 : 8;               // ISA 16-bit fragment K-octet select
    const _Float16* arow = m + ((size_t)b * T + nBase + rl) * (size_t)T;
    const _Float16* erow[MT];
    #pragma unroll
    for (int i = 0; i < MT; ++i)
        erow[i] = embT + ((size_t)b * D + dBase + i * 16 + rl) * (size_t)T;

    v8f acc[MT];
    #pragma unroll
    for (int i = 0; i < MT; ++i) acc[i] = (v8f){};

    for (int k0 = 0; k0 < T; k0 += 32) {
        // prefetch next A chunk (wrapped so the address is always in-bounds)
        __builtin_prefetch(arow + ((k0 + 1024) & (T - 1)), 0, 1);
        // A fragment: elements j<8 -> K=kb+j, j>=8 -> K=16+kb+(j-8)
        v8h a0 = *(const v8h*)(arow + k0 + kb);
        v8h a1 = *(const v8h*)(arow + k0 + 16 + kb);
        v16h a;
        #pragma unroll
        for (int j = 0; j < 8; ++j) { a[j] = a0[j]; a[j + 8] = a1[j]; }
        #pragma unroll
        for (int i = 0; i < MT; ++i) {
            v8h b0 = *(const v8h*)(erow[i] + k0 + kb);
            v8h b1 = *(const v8h*)(erow[i] + k0 + 16 + kb);
            v16h bf;
            #pragma unroll
            for (int j = 0; j < 8; ++j) { bf[j] = b0[j]; bf[j + 8] = b1[j]; }
            acc[i] = __builtin_amdgcn_wmma_f32_16x16x32_f16(
                false, a, false, bf, (short)0, acc[i], false, false);
        }
    }

    // blend: only node rows take the pooled average
    #pragma unroll
    for (int r = 0; r < 8; ++r) {
        int nrow = nBase + r + ((lane < 16) ? 0 : 8);
        if (pos[b * T + nrow] == 0) {
            float* o = out + ((size_t)b * T + nrow) * D + dBase + (lane & 15);
            #pragma unroll
            for (int i = 0; i < MT; ++i) o[i * 16] = acc[i][r];
        }
    }
}

extern "C" void kernel_launch(void* const* d_in, const int* in_sizes, int n_in,
                              void* d_out, int out_size, void* d_ws, size_t ws_size,
                              hipStream_t stream) {
    const int* code = (const int*)d_in[0];
    const int* pos  = (const int*)d_in[1];
    const unsigned char* attn = (const unsigned char*)d_in[2];
    const float* W  = (const float*)d_in[3];
    float* out = (float*)d_out;

    _Float16* embT = (_Float16*)((char*)d_ws + OFF_EMBT);
    _Float16* mbuf = (_Float16*)((char*)d_ws + OFF_M);

    k_embed<<<B * T, 192, 0, stream>>>(code, W, out);
    k_cnt_m<<<B * T, 256, 0, stream>>>(pos, attn, mbuf);
    k_transpose<<<dim3(D / 32, T / 32, B), 256, 0, stream>>>(out, embT);

    int tiles = B * NT * DG;                    // 6144 wave-tiles
    k_gemm<<<tiles / 8, 256, 0, stream>>>(pos, mbuf, embT, out);
}
